// GSAT_43087111913989
// MI455X (gfx1250) — matrix operand order repaired and checked by reading, not compiled, not run on hardware
//
#include <hip/hip_runtime.h>

#define BB 4
#define NN 2048
#define FD 64        // IN_F = HID = OUT_F
#define HD 256       // H * HID
#define ROWS (BB*NN) // 8192

typedef __attribute__((ext_vector_type(16))) __bf16 v16bf;
typedef __attribute__((ext_vector_type(8)))  __bf16 v8bf;
typedef __attribute__((ext_vector_type(4)))  __bf16 v4bf;
typedef __attribute__((ext_vector_type(8)))  float  v8f;
typedef __attribute__((ext_vector_type(2)))  int    v2i;

#if defined(__gfx1250__) && __has_builtin(__builtin_amdgcn_global_load_async_to_lds_b64) && __has_builtin(__builtin_amdgcn_s_wait_asynccnt)
#define HAVE_ASYNC_LDS 1
typedef __attribute__((address_space(1))) v2i gv2i;
typedef __attribute__((address_space(3))) v2i lv2i;
#else
#define HAVE_ASYNC_LDS 0
#endif

__device__ __forceinline__ __bf16 f2bf(float f) {
  union { float f; unsigned u; } x; x.f = f;
  unsigned r = x.u + 0x7fffu + ((x.u >> 16) & 1u);   // round-to-nearest-even
  union { unsigned short s; __bf16 b; } y; y.s = (unsigned short)(r >> 16);
  return y.b;
}

// ---------------- f32 -> bf16 flat convert ----------------
__global__ void cvt_bf16_kernel(const float* __restrict__ in, __bf16* __restrict__ out, int n) {
  int i = blockIdx.x * blockDim.x + threadIdx.x;
  if (i < n) out[i] = f2bf(in[i]);
}

// ---------------- ss/sd row dots: one wave per row ----------------
__global__ void rowdot_kernel(const float* __restrict__ h, const float* __restrict__ asrc,
                              const float* __restrict__ adst, float* __restrict__ ss,
                              float* __restrict__ sd) {
  int gw   = (int)((blockIdx.x * blockDim.x + threadIdx.x) >> 5);
  int lane = threadIdx.x & 31;
  if (gw >= ROWS) return;
  const float* r = h + (size_t)gw * HD;
  float vs = 0.f, vd = 0.f;
#pragma unroll
  for (int i = 0; i < 8; ++i) {
    int k = lane + i * 32;
    float hv = r[k];
    vs += hv * asrc[k];
    vd += hv * adst[k];
  }
#pragma unroll
  for (int off = 16; off > 0; off >>= 1) {
    vs += __shfl_xor(vs, off, 32);
    vd += __shfl_xor(vd, off, 32);
  }
  if (lane == 0) { ss[gw] = vs * 0.25f; sd[gw] = vd * 0.25f; }  // mean over H=4 heads
}

// ---------------- block reduction (256 threads, LDS) ----------------
__device__ __forceinline__ float block_reduce(float v, bool isMax) {
#pragma unroll
  for (int off = 16; off > 0; off >>= 1) {
    float o = __shfl_xor(v, off, 32);
    v = isMax ? fmaxf(v, o) : (v + o);
  }
  __shared__ float sm[8];
  int w = threadIdx.x >> 5, l = threadIdx.x & 31;
  __syncthreads();
  if (l == 0) sm[w] = v;
  __syncthreads();
  float r = sm[0];
#pragma unroll
  for (int i = 1; i < 8; ++i) r = isMax ? fmaxf(r, sm[i]) : (r + sm[i]);
  return r;
}

// ---------------- gumbel + double softmax, coalesced row-major att (bf16) ----------------
// one 256-thread block per (b, m) row; thread t owns columns [t*8, t*8+8)
__global__ void attn_softmax_kernel(const float* __restrict__ adj, const float* __restrict__ u,
                                    const float* __restrict__ ss, const float* __restrict__ sd,
                                    __bf16* __restrict__ att) {
  int bm = blockIdx.x;
  int b = bm >> 11, m = bm & 2047;
  const float* urow = u   + (size_t)bm * NN;
  const float* arow = adj + (size_t)m  * NN;
  const float* sdb  = sd  + (size_t)b  * NN;
  float sm = ss[bm];
  int base = threadIdx.x * 8;
  float4 uq0 = *(const float4*)(urow + base), uq1 = *(const float4*)(urow + base + 4);
  float4 aq0 = *(const float4*)(arow + base), aq1 = *(const float4*)(arow + base + 4);
  float4 dq0 = *(const float4*)(sdb  + base), dq1 = *(const float4*)(sdb  + base + 4);
  float uu[8] = {uq0.x, uq0.y, uq0.z, uq0.w, uq1.x, uq1.y, uq1.z, uq1.w};
  float aa[8] = {aq0.x, aq0.y, aq0.z, aq0.w, aq1.x, aq1.y, aq1.z, aq1.w};
  float dd[8] = {dq0.x, dq0.y, dq0.z, dq0.w, dq1.x, dq1.y, dq1.z, dq1.w};
  float lg[8];
  float mx = -3.402823466e38f;
#pragma unroll
  for (int i = 0; i < 8; ++i) {
    float g = -__logf(-__logf(uu[i] + 1e-9f) + 1e-9f);  // gumbel noise
    lg[i] = (sm + dd[i]) * aa[i] + g;                   // TAU = 1
    mx = fmaxf(mx, lg[i]);
  }
  mx = block_reduce(mx, true);
  float s = 0.f;
#pragma unroll
  for (int i = 0; i < 8; ++i) { lg[i] = __expf(lg[i] - mx); s += lg[i]; }
  s = block_reduce(s, false);
  float inv = 1.0f / s;   // == max(att): exp is monotone, max element is exp(0)/s
  float s2 = 0.f;
#pragma unroll
  for (int i = 0; i < 8; ++i) { lg[i] = __expf(lg[i] * inv - inv); s2 += lg[i]; }
  s2 = block_reduce(s2, false);
  float r2 = 1.0f / s2;
  v8bf o;
#pragma unroll
  for (int i = 0; i < 8; ++i) o[i] = f2bf(lg[i] * r2);
  *(v8bf*)(att + (size_t)bm * NN + base) = o;           // 16B coalesced store
}

// ---------------- att[b][m][n] bf16 -> attT[b][n][m] bf16, LDS 32x32 tiles ----------------
// async global->LDS copy on toolchains exposing the gfx1250 async builtins
__global__ void att_transpose_kernel(const __bf16* __restrict__ att, __bf16* __restrict__ attT) {
  const int tpr = NN / 32;                // 64 tiles per row
  int tr = blockIdx.x / tpr;              // m-tile
  int tc = blockIdx.x - tr * tpr;         // n-tile
  const __bf16* src = att  + (size_t)blockIdx.y * NN * NN + (size_t)(tr * 32) * NN + tc * 32;
  __bf16*       dst = attT + (size_t)blockIdx.y * NN * NN + (size_t)(tc * 32) * NN + tr * 32;
  __shared__ __bf16 tile[32][40];         // pad keeps 8B alignment, breaks bank conflicts
  int t = threadIdx.x;
  int r = t >> 3;                         // 0..31 row within tile
  int c = (t & 7) * 4;                    // 0..28 col chunk
#if HAVE_ASYNC_LDS
  __builtin_amdgcn_global_load_async_to_lds_b64(
      (gv2i*)(v2i*)(src + (size_t)r * NN + c),
      (lv2i*)(v2i*)&tile[r][c], 0, 0);
  __builtin_amdgcn_s_wait_asynccnt(0);
#else
  *(v4bf*)&tile[r][c] = *(const v4bf*)(src + (size_t)r * NN + c);
#endif
  __syncthreads();
  v4bf o;
#pragma unroll
  for (int i = 0; i < 4; ++i) o[i] = tile[c + i][r];
  *(v4bf*)(dst + (size_t)r * NN + c) = o;               // coalesced 8B store
}

// ---------------- h[b][m][256] f32 -> HT[b][256][2048] bf16, LDS 32x32 tiles ----------------
__global__ void h_transpose_kernel(const float* __restrict__ h, __bf16* __restrict__ ht) {
  const int tpr = HD / 32;                // 8 tiles per row
  int tr = blockIdx.x / tpr;              // m-tile (0..63)
  int tc = blockIdx.x - tr * tpr;         // j-tile (0..7)
  const float* src = h  + (size_t)blockIdx.y * NN * HD + (size_t)(tr * 32) * HD + tc * 32;
  __bf16*      dst = ht + (size_t)blockIdx.y * HD * NN + (size_t)(tc * 32) * NN + tr * 32;
  __shared__ __bf16 tile[32][40];
  int t = threadIdx.x;
  int r = t >> 3;
  int c = (t & 7) * 4;
  float4 v = *(const float4*)(src + (size_t)r * HD + c);
  tile[r][c + 0] = f2bf(v.x); tile[r][c + 1] = f2bf(v.y);
  tile[r][c + 2] = f2bf(v.z); tile[r][c + 3] = f2bf(v.w);
  __syncthreads();
  v4bf o;
#pragma unroll
  for (int i = 0; i < 4; ++i) o[i] = tile[c + i][r];
  *(v4bf*)(dst + (size_t)r * NN + c) = o;
}

// ---------------- WMMA fragment loaders (ISA 7.12.2 layouts) ----------------
__device__ __forceinline__ v16bf loadA(const __bf16* p, int k, int sel) {
  v8bf lo = *(const v8bf*)(p + k + sel * 8);
  v8bf hi = *(const v8bf*)(p + k + 16 + sel * 8);
  v16bf v;
#pragma unroll
  for (int i = 0; i < 8; ++i) { v[i] = lo[i]; v[i + 8] = hi[i]; }
  return v;
}
__device__ __forceinline__ v16bf loadB(const __bf16* p, int k, int sel) {
  v8bf lo = *(const v8bf*)(p + k + sel * 8);
  v8bf hi = *(const v8bf*)(p + k + 16 + sel * 8);
  v16bf v;
#pragma unroll
  for (int i = 0; i < 8; ++i) { v[i] = lo[i]; v[i + 8] = hi[i]; }
  return v;
}

// ---------------- bf16 WMMA GEMM: C[M,N] = A[M,K] * B'[N,K]^T ----------------
// one wave per 32x32 tile (2x2 register blocking => 4 WMMA per 8 b128 loads)
__global__ void gemm_wmma_kernel(const __bf16* __restrict__ A, const __bf16* __restrict__ Bm,
                                 float* __restrict__ C, int Mt2, int Nt2, int K,
                                 long long sA, long long sB, long long sC) {
  int wave  = (int)((blockIdx.x * blockDim.x + threadIdx.x) >> 5);
  int tiles = Mt2 * Nt2;
  if (wave >= tiles) return;             // whole-wave exit keeps EXEC all-ones for WMMA
  int tm = wave / Nt2, tn = wave - tm * Nt2;
  int lane = threadIdx.x & 31;
  int r = lane & 15, sel = lane >> 4;
  const __bf16* Ab0 = A  + (size_t)blockIdx.y * sA + (size_t)(tm * 32 + r) * K;
  const __bf16* Ab1 = Ab0 + (size_t)16 * K;
  const __bf16* Bb0 = Bm + (size_t)blockIdx.y * sB + (size_t)(tn * 32 + r) * K;
  const __bf16* Bb1 = Bb0 + (size_t)16 * K;
  v8f acc00 = {0.f,0.f,0.f,0.f,0.f,0.f,0.f,0.f};
  v8f acc01 = acc00, acc10 = acc00, acc11 = acc00;
  for (int k = 0; k < K; k += 32) {
    v16bf a0 = loadA(Ab0, k, sel);
    v16bf a1 = loadA(Ab1, k, sel);
    v16bf b0 = loadB(Bb0, k, sel);
    v16bf b1 = loadB(Bb1, k, sel);
    acc00 = __builtin_amdgcn_wmma_f32_16x16x32_bf16(false, a0, false, b0, (short)0, acc00, false, false);
    acc01 = __builtin_amdgcn_wmma_f32_16x16x32_bf16(false, a0, false, b1, (short)0, acc01, false, false);
    acc10 = __builtin_amdgcn_wmma_f32_16x16x32_bf16(false, a1, false, b0, (short)0, acc10, false, false);
    acc11 = __builtin_amdgcn_wmma_f32_16x16x32_bf16(false, a1, false, b1, (short)0, acc11, false, false);
  }
  int ldc = Nt2 * 32;
  // D layout: element e -> D[sel*8 + e, lane&15]
  float* Cb = C + (size_t)blockIdx.y * sC + (size_t)(tm * 32 + sel * 8) * ldc + tn * 32 + r;
#pragma unroll
  for (int e = 0; e < 8; ++e) {
    Cb[(size_t)e * ldc]            = acc00[e];
    Cb[(size_t)e * ldc + 16]       = acc01[e];
    Cb[(size_t)(e + 16) * ldc]     = acc10[e];
    Cb[(size_t)(e + 16) * ldc + 16]= acc11[e];
  }
}

extern "C" void kernel_launch(void* const* d_in, const int* in_sizes, int n_in,
                              void* d_out, int out_size, void* d_ws, size_t ws_size,
                              hipStream_t stream) {
  (void)in_sizes; (void)n_in; (void)out_size; (void)ws_size;
  const float* x   = (const float*)d_in[0];
  const float* adj = (const float*)d_in[1];
  const float* u1  = (const float*)d_in[2];
  const float* u2  = (const float*)d_in[3];
  const float* W1  = (const float*)d_in[4];
  const float* as1 = (const float*)d_in[5];
  const float* ad1 = (const float*)d_in[6];
  const float* W2  = (const float*)d_in[7];
  const float* as2 = (const float*)d_in[8];
  const float* ad2 = (const float*)d_in[9];
  const float* O1  = (const float*)d_in[10];
  const float* O2  = (const float*)d_in[11];
  float* out = (float*)d_out;

  char* ws = (char*)d_ws;
  size_t off = 0;
  auto take = [&](size_t bytes) -> char* {
    char* p = ws + off; off = (off + bytes + 255) & ~(size_t)255; return p;
  };
  __bf16* xbf  = (__bf16*)take((size_t)ROWS * FD * 2);
  __bf16* w1bf = (__bf16*)take((size_t)HD * FD * 2);
  __bf16* w2bf = (__bf16*)take((size_t)HD * FD * 2);
  __bf16* o1bf = (__bf16*)take((size_t)FD * HD * 2);
  __bf16* o2bf = (__bf16*)take((size_t)FD * HD * 2);
  float*  ssb  = (float*) take((size_t)ROWS * 4);
  float*  sdb  = (float*) take((size_t)ROWS * 4);
  float*  hbuf = (float*) take((size_t)ROWS * HD * 4);
  __bf16* htbf = (__bf16*)take((size_t)BB * HD * NN * 2);
  __bf16* attT = (__bf16*)take((size_t)BB * NN * NN * 2);
  // region R: row-major att (32MB) time-multiplexed with agg/aggbf/hp/hpbf (15MB)
  char*   R    = take((size_t)BB * NN * NN * 2);
  __bf16* attbuf = (__bf16*)R;
  float*  agg    = (float*) R;
  __bf16* aggbf  = (__bf16*)(R + (size_t)ROWS * HD * 4);
  float*  hp     = (float*) (R + (size_t)ROWS * HD * 4 + (size_t)ROWS * HD * 2);
  __bf16* hpbf   = (__bf16*)(R + (size_t)ROWS * HD * 4 + (size_t)ROWS * HD * 2 + (size_t)ROWS * FD * 4);

  auto cvt = [&](const float* in, __bf16* o, int n) {
    cvt_bf16_kernel<<<(n + 255) / 256, 256, 0, stream>>>(in, o, n);
  };
  auto gemm = [&](const __bf16* A, const __bf16* Bm, float* C, int M, int N, int K,
                  long long sA, long long sB, long long sC, int batches) {
    int tiles = (M / 32) * (N / 32);
    dim3 grid((tiles * 32 + 255) / 256, batches);
    gemm_wmma_kernel<<<grid, 256, 0, stream>>>(A, Bm, C, M / 32, N / 32, K, sA, sB, sC);
  };

  // operand conversions (weights already stored as B' = [N,K])
  cvt(x,  xbf,  ROWS * FD);
  cvt(W1, w1bf, HD * FD);
  cvt(W2, w2bf, HD * FD);
  cvt(O1, o1bf, FD * HD);
  cvt(O2, o2bf, FD * HD);

  const long long sAtt = (long long)NN * NN, sHT = (long long)HD * NN, sAgg = (long long)NN * HD;

  // ---- stage 1 ----
  gemm(xbf, w1bf, hbuf, ROWS, HD, FD, 0, 0, 0, 1);                        // h1 = x @ W1^T
  rowdot_kernel<<<ROWS * 32 / 256, 256, 0, stream>>>(hbuf, as1, ad1, ssb, sdb);
  h_transpose_kernel<<<dim3((NN/32)*(HD/32), BB), 256, 0, stream>>>(hbuf, htbf);
  attn_softmax_kernel<<<ROWS, 256, 0, stream>>>(adj, u1, ssb, sdb, attbuf);
  att_transpose_kernel<<<dim3((NN/32)*(NN/32), BB), 256, 0, stream>>>(attbuf, attT);
  gemm(attT, htbf, agg, NN, HD, NN, sAtt, sHT, sAgg, BB);                 // att^T @ h1
  cvt(agg, aggbf, ROWS * HD);
  gemm(aggbf, o1bf, hp, ROWS, FD, HD, 0, 0, 0, 1);                        // hp1 = agg @ O1^T
  cvt(hp, hpbf, ROWS * FD);

  // ---- stage 2 ----
  gemm(hpbf, w2bf, hbuf, ROWS, HD, FD, 0, 0, 0, 1);                       // h2 = hp1 @ W2^T
  rowdot_kernel<<<ROWS * 32 / 256, 256, 0, stream>>>(hbuf, as2, ad2, ssb, sdb);
  h_transpose_kernel<<<dim3((NN/32)*(HD/32), BB), 256, 0, stream>>>(hbuf, htbf);
  attn_softmax_kernel<<<ROWS, 256, 0, stream>>>(adj, u2, ssb, sdb, attbuf);
  att_transpose_kernel<<<dim3((NN/32)*(NN/32), BB), 256, 0, stream>>>(attbuf, attT);
  gemm(attT, htbf, agg, NN, HD, NN, sAtt, sHT, sAgg, BB);                 // att^T @ h2
  cvt(agg, aggbf, ROWS * HD);
  gemm(aggbf, o2bf, out, ROWS, FD, HD, 0, 0, 0, 1);                       // out = agg @ O2^T
}
